// SocialLSTM_6141803233972
// MI455X (gfx1250) — matrix operand order repaired and checked by reading, not compiled, not run on hardware
//
#include <hip/hip_runtime.h>
#include <hip/hip_bf16.h>

#define SEQ 4096
#define EMBD 64
#define HID 32

typedef __attribute__((ext_vector_type(16))) _Float16 v16h;
typedef __attribute__((ext_vector_type(8)))  _Float16 v8h;
typedef __attribute__((ext_vector_type(8)))  float    v8f;

// ---- LDS layout (bytes). Weight rows padded to 104 halves (208B = 52 dwords)
// so the 16 lanes of a B-fragment ds_load_b128 hit distinct banks.
#define WSTRIDE   104
#define OFF_WENC  0
#define OFF_WDEC  (128 * WSTRIDE * 2)              // 26624
#define OFF_XVEC  (OFF_WDEC + 128 * WSTRIDE * 2)   // 53248  x = [emb(64) f16 ; h(32) f16]
#define OFF_ZERO  (OFF_XVEC + 256)                 // 53504  64B zeros for inactive A lanes
#define LDS_BYTES (OFF_ZERO + 64)                  // 53568

// Single-wave workgroup: DS ops from one wave execute in order (ISA ch.11), so a
// same-address store->load needs no DScnt drain -- only a compiler reorder fence.
__device__ __forceinline__ void lds_fence() { asm volatile("" ::: "memory"); }

__device__ __forceinline__ float sigf(float x) {
    return 1.0f / (1.0f + __expf(-x));
}
__device__ __forceinline__ float tanh_fast(float x) {
    x = fminf(fmaxf(x, -12.0f), 12.0f);
    float e = __expf(2.0f * x);
    return (e - 1.0f) / (e + 1.0f);
}

// A fragment (16x32 f16), row 0 = x[k0..k0+31], rows 1..15 = 0.
// ISA layout: lane0 holds row0 K {0..7,16..23}, lane16 holds K {8..15,24..31}.
__device__ __forceinline__ v16h load_afrag(const char* lds, int lane, int k0) {
    unsigned base;
    if (lane == 0)       base = OFF_XVEC + 2u * k0;
    else if (lane == 16) base = OFF_XVEC + 2u * k0 + 16u;
    else                 base = OFF_ZERO;
    v8h lo = *(const v8h*)(lds + base);
    v8h hi = *(const v8h*)(lds + base + 32);
    return __builtin_shufflevector(lo, hi, 0,1,2,3,4,5,6,7,8,9,10,11,12,13,14,15);
}

// B fragment = W^T tile: B[k,n] = W[g0+n, k0+k].  Lane n holds col n%16,
// K (n<16 ? {0..7,16..23} : {8..15,24..31}) from the padded W row.
__device__ __forceinline__ v16h load_bfrag(const char* lds, unsigned woff,
                                           int lane, int g0, int k0) {
    unsigned addr = woff + (unsigned)((g0 + (lane & 15)) * (WSTRIDE * 2)
                                      + k0 * 2 + ((lane >> 4) << 4));
    v8h lo = *(const v8h*)(lds + addr);
    v8h hi = *(const v8h*)(lds + addr + 32);
    return __builtin_shufflevector(lo, hi, 0,1,2,3,4,5,6,7,8,9,10,11,12,13,14,15);
}

// Hoist the loop-invariant Whh (K-chunk 2) B-fragments into registers.
__device__ __forceinline__ void load_bh(const char* lds, unsigned woff, int lane,
                                        v16h Bh[8]) {
#pragma unroll
    for (int t = 0; t < 8; ++t) Bh[t] = load_bfrag(lds, woff, lane, 16 * t, 64);
}

// One LSTM cell update: 24 WMMAs (8 gate tiles x 3 K-chunks), C starts as the
// inline-zero operand; bias folded in with one scalar add on component 0.
__device__ __forceinline__ void cell_step(char* lds, int lane, unsigned woff,
                                          const v16h* Bh, const float* biasv,
                                          float& c_lo, float& c_hi,
                                          float& h_lo, float& h_hi) {
    v16h A0 = load_afrag(lds, lane, 0);
    v16h A1 = load_afrag(lds, lane, 32);
    v16h A2 = load_afrag(lds, lane, 64);
    float g[8];
#pragma unroll
    for (int t = 0; t < 8; ++t) {
        v16h B0 = load_bfrag(lds, woff, lane, 16 * t, 0);
        v16h B1 = load_bfrag(lds, woff, lane, 16 * t, 32);
        v8f acc = (v8f){0.f, 0.f, 0.f, 0.f, 0.f, 0.f, 0.f, 0.f};   // inline 0 SRC2
        acc = __builtin_amdgcn_wmma_f32_16x16x32_f16(false, A0, false, B0, (short)0, acc, false, false);
        acc = __builtin_amdgcn_wmma_f32_16x16x32_f16(false, A1, false, B1, (short)0, acc, false, false);
        acc = __builtin_amdgcn_wmma_f32_16x16x32_f16(false, A2, false, Bh[t], (short)0, acc, false, false);
        g[t] = acc[0] + biasv[t];                        // D[0, lane], lanes 0..15
    }
    // PyTorch gate order i,f,g,o; lanes 0..15 hold channels {lane, lane+16}.
    float i0 = sigf(g[0]),       i1 = sigf(g[1]);
    float f0 = sigf(g[2]),       f1 = sigf(g[3]);
    float gg0 = tanh_fast(g[4]), gg1 = tanh_fast(g[5]);
    float o0 = sigf(g[6]),       o1 = sigf(g[7]);
    c_lo = f0 * c_lo + i0 * gg0;
    c_hi = f1 * c_hi + i1 * gg1;
    h_lo = o0 * tanh_fast(c_lo);
    h_hi = o1 * tanh_fast(c_hi);
    if (lane < 16) {                                     // h (f16) -> next step's A2
        _Float16* xh = (_Float16*)(lds + OFF_XVEC + 2 * EMBD);
        xh[lane]      = (_Float16)h_lo;
        xh[lane + 16] = (_Float16)h_hi;
    }
    lds_fence();
}

// vel = W_out(2x32) @ h + b_out. h is register-resident; butterfly reduction.
__device__ __forceinline__ float2 compute_vel(int lane, float h_lo, float h_hi,
                                              float wo0, float wo1,
                                              float bo0, float bo1) {
    float hh = __shfl_up(h_hi, 16, 32);                  // lanes 16..31 <- h_hi[lane-16]
    float hv = (lane < 16) ? h_lo : hh;
    float p0 = wo0 * hv;
    float p1 = wo1 * hv;
#pragma unroll
    for (int m = 16; m > 0; m >>= 1) {
        p0 += __shfl_xor(p0, m, 32);
        p1 += __shfl_xor(p1, m, 32);
    }
    return make_float2(p0 + bo0, p1 + bo1);
}

// emb = relu(W_emb @ vel + b_emb) -> xvec[0..63] (f16); 2 channels per lane.
__device__ __forceinline__ void compute_emb(char* lds, int lane, float vx, float vy,
                                            float we0, float we1, float we2,
                                            float we3, float be0, float be1) {
    _Float16* xv = (_Float16*)(lds + OFF_XVEC);
    float e0 = fmaxf(we0 * vx + we1 * vy + be0, 0.0f);
    float e1 = fmaxf(we2 * vx + we3 * vy + be1, 0.0f);
    xv[lane]      = (_Float16)e0;
    xv[lane + 32] = (_Float16)e1;
    lds_fence();
}

__global__ void __launch_bounds__(32)
social_lstm_kernel(const float* __restrict__ obs, const float* __restrict__ W_emb,
                   const float* __restrict__ b_emb,
                   const float* __restrict__ eWih, const float* __restrict__ eWhh,
                   const float* __restrict__ ebih, const float* __restrict__ ebhh,
                   const float* __restrict__ dWih, const float* __restrict__ dWhh,
                   const float* __restrict__ dbih, const float* __restrict__ dbhh,
                   const float* __restrict__ W_out, const float* __restrict__ b_out,
                   const int* __restrict__ np_p, float* __restrict__ out) {
    __shared__ float4 lds4[LDS_BYTES / 16];
    char* lds = (char*)lds4;
    const int lane = threadIdx.x;
    const int NP = np_p[0];

    // ---- one-time weight staging: f32 -> f16 [128][104] row-padded, W = [Wih | Whh]
    _Float16* We = (_Float16*)(lds + OFF_WENC);
    _Float16* Wd = (_Float16*)(lds + OFF_WDEC);
    for (int i = lane; i < 128 * 64; i += 32) {
        int r = i >> 6, k = i & 63;
        We[r * WSTRIDE + k] = (_Float16)eWih[i];
        Wd[r * WSTRIDE + k] = (_Float16)dWih[i];
    }
    for (int i = lane; i < 128 * 32; i += 32) {
        int r = i >> 5, k = i & 31;
        We[r * WSTRIDE + 64 + k] = (_Float16)eWhh[i];
        Wd[r * WSTRIDE + 64 + k] = (_Float16)dWhh[i];
    }
    if (lane < 16) ((float*)(lds + OFF_ZERO))[lane] = 0.0f;
    {   // x vector + h state start at zero
        _Float16* xv = (_Float16*)(lds + OFF_XVEC);
        for (int i = lane; i < 96; i += 32) xv[i] = (_Float16)0.0f;
    }
    __syncthreads();

    // ---- hoist small loop-invariants straight from global into registers
    const float we0 = W_emb[2 * lane],            we1 = W_emb[2 * lane + 1];
    const float we2 = W_emb[2 * (lane + 32)],     we3 = W_emb[2 * (lane + 32) + 1];
    const float be0 = b_emb[lane],                be1 = b_emb[lane + 32];
    const float wo0 = W_out[lane],                wo1 = W_out[32 + lane];
    const float bo0 = b_out[0],                   bo1 = b_out[1];
    const int   l15 = lane & 15;
    float biasE[8], biasD[8];
#pragma unroll
    for (int t = 0; t < 8; ++t) {
        biasE[t] = ebih[16 * t + l15] + ebhh[16 * t + l15];
        biasD[t] = dbih[16 * t + l15] + dbhh[16 * t + l15];
    }

    float c_lo = 0.f, c_hi = 0.f, h_lo = 0.f, h_hi = 0.f;

    const float2* obs2 = (const float2*)obs;
    float2 vcur = obs2[0];

    // ---------------- encoder: 4096 serial steps ----------------
    {
        v16h BhE[8];
        load_bh(lds, OFF_WENC, lane, BhE);               // Whh fragments stay in VGPRs
#pragma unroll 1
        for (int t = 0; t < SEQ; ++t) {
            int tn = (t + 1 < SEQ) ? t + 1 : SEQ - 1;
            int tp = (t + 16 < SEQ) ? t + 16 : SEQ - 1;
            __builtin_prefetch(&obs2[tp], 0, 1);         // global_prefetch, warm caches
            float2 vnext = obs2[tn];                     // 1-step software pipeline
            compute_emb(lds, lane, vcur.x, vcur.y, we0, we1, we2, we3, be0, be1);
            cell_step(lds, lane, OFF_WENC, BhE, biasE, c_lo, c_hi, h_lo, h_hi);
            float2 v = compute_vel(lane, h_lo, h_hi, wo0, wo1, bo0, bo1);
            if (lane == 0) { out[2 * t] = v.x; out[2 * t + 1] = v.y; }
            vcur = vnext;
        }
    }

    // ---------------- decoder ----------------
    {
        v16h BhD[8];
        load_bh(lds, OFF_WDEC, lane, BhD);
        // seed: start_tag (one-hot at EMB-1)
        _Float16* xv = (_Float16*)(lds + OFF_XVEC);
        for (int i = lane; i < 64; i += 32) xv[i] = (_Float16)((i == 63) ? 1.0f : 0.0f);
        lds_fence();
        cell_step(lds, lane, OFF_WDEC, BhD, biasD, c_lo, c_hi, h_lo, h_hi);

#pragma unroll 1
        for (int k = 0; k < NP; ++k) {
            float2 v = compute_vel(lane, h_lo, h_hi, wo0, wo1, bo0, bo1);
            if (lane == 0) { out[2 * (SEQ + k)] = v.x; out[2 * (SEQ + k) + 1] = v.y; }
            compute_emb(lds, lane, v.x, v.y, we0, we1, we2, we3, be0, be1);
            cell_step(lds, lane, OFF_WDEC, BhD, biasD, c_lo, c_hi, h_lo, h_hi);
        }
    }
}

extern "C" void kernel_launch(void* const* d_in, const int* in_sizes, int n_in,
                              void* d_out, int out_size, void* d_ws, size_t ws_size,
                              hipStream_t stream) {
    (void)in_sizes; (void)n_in; (void)out_size; (void)d_ws; (void)ws_size;
    social_lstm_kernel<<<1, 32, 0, stream>>>(
        (const float*)d_in[0],  (const float*)d_in[1],  (const float*)d_in[2],
        (const float*)d_in[3],  (const float*)d_in[4],  (const float*)d_in[5],
        (const float*)d_in[6],  (const float*)d_in[7],  (const float*)d_in[8],
        (const float*)d_in[9],  (const float*)d_in[10], (const float*)d_in[11],
        (const float*)d_in[12], (const int*)d_in[13],
        (float*)d_out);
}
